// mymodel_6674379178666
// MI455X (gfx1250) — matrix operand order repaired and checked by reading, not compiled, not run on hardware
//
#include <hip/hip_runtime.h>
#include <stdint.h>

// ---------------------------------------------------------------------------
// Types for CDNA5 WMMA
// ---------------------------------------------------------------------------
typedef __attribute__((ext_vector_type(16))) __bf16 v16bf;
typedef __attribute__((ext_vector_type(8)))  float  v8f;

union FragBF {
    v16bf v;
    uint4 u[2];
};

__device__ __forceinline__ __bf16 f2bf(float f) {
    union { float f; uint32_t u; } c; c.f = f;
    uint32_t u = c.u;
    u += 0x7FFFu + ((u >> 16) & 1u);           // round-to-nearest-even
    union { unsigned short s; __bf16 b; } o;
    o.s = (unsigned short)(u >> 16);
    return o.b;
}

// ---------------------------------------------------------------------------
// Edge kernels: Monte-Carlo conv gather/scatter
// agg layout: agg[d][h][c] = agg[d*(16*Cin) + h*Cin + c]  (== reshape(M, H*Cin))
// ---------------------------------------------------------------------------
__global__ void edge_count_kernel(const int* __restrict__ dst, float* __restrict__ cnt, int E) {
    int e = blockIdx.x * blockDim.x + threadIdx.x;
    if (e >= E) return;
    atomicAdd(&cnt[dst[e]], 1.0f);
}

__global__ void edge_scatter_kernel(const float* __restrict__ feat,     // [S, Cin]
                                    const float* __restrict__ pos_src,  // [S, 3]
                                    const float* __restrict__ pos_dst,  // [M, 3]
                                    const int*   __restrict__ src,
                                    const int*   __restrict__ dst,
                                    const float* __restrict__ Wb,       // [3, 16]
                                    const float* __restrict__ bb,       // [16]
                                    float inv_r, int E, int Cin,
                                    float* __restrict__ agg)            // [M, 16*Cin]
{
    long long tid   = (long long)blockIdx.x * blockDim.x + threadIdx.x;
    long long total = (long long)E * Cin;
    if (tid >= total) return;
    int e = (int)(tid / Cin);
    int c = (int)(tid % Cin);
    int s = src[e];
    int d = dst[e];

    float dx = (pos_src[3 * s + 0] - pos_dst[3 * d + 0]) * inv_r;
    float dy = (pos_src[3 * s + 1] - pos_dst[3 * d + 1]) * inv_r;
    float dz = (pos_src[3 * s + 2] - pos_dst[3 * d + 2]) * inv_r;
    float fv = feat[(size_t)s * Cin + c];

    float* out = agg + (size_t)d * (16 * Cin) + c;
#pragma unroll
    for (int hh = 0; hh < 16; ++hh) {
        float h = fmaf(dx, Wb[hh], fmaf(dy, Wb[16 + hh], fmaf(dz, Wb[32 + hh], bb[hh])));
        h = (h > 0.0f) ? h : 0.2f * h;        // leaky_relu basis
        atomicAdd(out + hh * Cin, h * fv);
    }
}

// ---------------------------------------------------------------------------
// Normalize by neighbor count and pack to bf16, K padded to multiple of 32.
// cnt == nullptr => no division (used to pack plain activations for GEMM).
// ---------------------------------------------------------------------------
__global__ void norm_pack_kernel(const float* __restrict__ agg, const float* __restrict__ cnt,
                                 __bf16* __restrict__ out, int M, int K, int Kpad)
{
    long long tid = (long long)blockIdx.x * blockDim.x + threadIdx.x;
    if (tid >= (long long)M * Kpad) return;
    int m  = (int)(tid / Kpad);
    int kk = (int)(tid % Kpad);
    float v = 0.0f;
    if (kk < K) {
        v = agg[(size_t)m * K + kk];
        if (cnt) {
            float c = cnt[m];
            v /= (c < 1.0f ? 1.0f : c);
        }
    }
    out[(size_t)m * Kpad + kk] = f2bf(v);
}

// W [K, N] f32 row-major  ->  WT [N, Kpad] bf16 (zero-padded K)
__global__ void pack_wt_kernel(const float* __restrict__ W, __bf16* __restrict__ WT,
                               int K, int N, int Kpad)
{
    long long tid = (long long)blockIdx.x * blockDim.x + threadIdx.x;
    if (tid >= (long long)N * Kpad) return;
    int n = (int)(tid / Kpad);
    int k = (int)(tid % Kpad);
    WT[(size_t)n * Kpad + k] = f2bf(k < K ? W[(size_t)k * N + n] : 0.0f);
}

// ---------------------------------------------------------------------------
// WMMA GEMM: C[M,N] f32 = A[M,Kpad] bf16 @ BT[N,Kpad]^T bf16  (+ optional bias)
// One wave per 16x16 tile; K-loop step 32 with v_wmma_f32_16x16x32_bf16.
// M-tail handling: clamp the A row (garbage rows only feed D rows >= M,
// which the store loop discards) -> no per-iteration EXEC masking/zero-fill.
// ---------------------------------------------------------------------------
__global__ void gemm_wmma_kernel(const __bf16* __restrict__ A,
                                 const __bf16* __restrict__ BT,
                                 const float*  __restrict__ bias,
                                 float* __restrict__ C,
                                 int M, int N, int Kpad)
{
    int wave = (blockIdx.x * blockDim.x + threadIdx.x) >> 5;
    int lane = threadIdx.x & 31;
    int tilesM = (M + 15) >> 4;
    int tilesN = N >> 4;
    if (wave >= tilesM * tilesN) return;

    int tm = wave % tilesM;
    int tn = wave / tilesM;
    int m0 = tm << 4;
    int n0 = tn << 4;
    int r  = lane & 15;
    int hi = lane >> 4;

    int rowA = m0 + r;
    if (rowA >= M) rowA = M - 1;                 // clamp: contributes only to discarded D rows
    const __bf16* arow = A  + (size_t)rowA * Kpad;
    const __bf16* brow = BT + (size_t)(n0 + r) * Kpad;

    v8f acc = {};

    for (int k0 = 0; k0 < Kpad; k0 += 32) {
        FragBF a, b;
        // 16-bit A 16x32 layout: halves 0..7 -> K = hi*8 + 0..7,
        //                        halves 8..15 -> K = 16 + hi*8 + 0..7
        a.u[0] = *(const uint4*)(arow + k0 + hi * 8);
        a.u[1] = *(const uint4*)(arow + k0 + 16 + hi * 8);
        // 16-bit B 32x16 layout: lane holds column n0+r, halves 0..7 -> K = hi*8 + 0..7,
        //                        halves 8..15 -> K = 16 + hi*8 + 0..7
        b.u[0] = *(const uint4*)(brow + k0 + hi * 8);
        b.u[1] = *(const uint4*)(brow + k0 + 16 + hi * 8);

        if (k0 + 32 < Kpad) {
            __builtin_prefetch(arow + k0 + 32, 0, 1);   // global_prefetch_b8
            __builtin_prefetch(brow + k0 + 32, 0, 1);
        }

        acc = __builtin_amdgcn_wmma_f32_16x16x32_bf16(
                  /*neg_a=*/false, a.v, /*neg_b=*/false, b.v,
                  /*c_mod=*/(short)0, acc, /*reuse_a=*/false, /*reuse_b=*/false);
    }

    int   col    = n0 + r;
    float bias_v = bias ? bias[col] : 0.0f;
#pragma unroll
    for (int v = 0; v < 8; ++v) {
        int row = m0 + hi * 8 + v;     // C/D layout: VGPR v -> M = v (+8 for hi lanes)
        if (row < M) {
            C[(size_t)row * N + col] = acc[v] + bias_v;
        }
    }
}

// ---------------------------------------------------------------------------
// BatchNorm (training-mode batch stats) + leaky_relu(0.3)
// ---------------------------------------------------------------------------
__global__ void col_stats_kernel(const float* __restrict__ X, int M, int N,
                                 float* __restrict__ sum, float* __restrict__ sumsq)
{
    int c = blockIdx.x;                 // one block per column
    float s = 0.0f, q = 0.0f;
    for (int m = threadIdx.x; m < M; m += blockDim.x) {
        float v = X[(size_t)m * N + c];
        s += v;
        q += v * v;
    }
    __shared__ float ss[256];
    __shared__ float qq[256];
    ss[threadIdx.x] = s;
    qq[threadIdx.x] = q;
    __syncthreads();
    for (int o = 128; o > 0; o >>= 1) {
        if ((int)threadIdx.x < o) {
            ss[threadIdx.x] += ss[threadIdx.x + o];
            qq[threadIdx.x] += qq[threadIdx.x + o];
        }
        __syncthreads();
    }
    if (threadIdx.x == 0) { sum[c] = ss[0]; sumsq[c] = qq[0]; }
}

__global__ void bn_lrelu_kernel(float* __restrict__ X, int M, int N,
                                const float* __restrict__ sum, const float* __restrict__ sumsq,
                                const float* __restrict__ g, const float* __restrict__ b)
{
    long long tid = (long long)blockIdx.x * blockDim.x + threadIdx.x;
    if (tid >= (long long)M * N) return;
    int c = (int)(tid % N);
    float invM  = 1.0f / (float)M;
    float mean  = sum[c] * invM;
    float var   = sumsq[c] * invM - mean * mean;
    float y     = (X[tid] - mean) * rsqrtf(var + 1e-3f) * g[c] + b[c];
    X[tid] = (y > 0.0f) ? y : 0.3f * y;
}

// ---------------------------------------------------------------------------
// Host-side orchestration
// ---------------------------------------------------------------------------
static inline unsigned cdiv(long long a, long long b) { return (unsigned)((a + b - 1) / b); }

extern "C" void kernel_launch(void* const* d_in, const int* in_sizes, int n_in,
                              void* d_out, int out_size, void* d_ws, size_t ws_size,
                              hipStream_t stream)
{
    (void)n_in; (void)out_size; (void)ws_size;

    const float* pos[4]  = { (const float*)d_in[0], (const float*)d_in[1],
                             (const float*)d_in[2], (const float*)d_in[3] };
    const int*   src[3]  = { (const int*)d_in[4], (const int*)d_in[6], (const int*)d_in[8] };
    const int*   dstm[3] = { (const int*)d_in[5], (const int*)d_in[7], (const int*)d_in[9] };
    const float* feat0   = (const float*)d_in[10];
    const float* Wb[3]   = { (const float*)d_in[11], (const float*)d_in[16], (const float*)d_in[21] };
    const float* bb[3]   = { (const float*)d_in[12], (const float*)d_in[17], (const float*)d_in[22] };
    const float* Wc[3]   = { (const float*)d_in[13], (const float*)d_in[18], (const float*)d_in[23] };
    const float* gg[3]   = { (const float*)d_in[14], (const float*)d_in[19], (const float*)d_in[24] };
    const float* be[3]   = { (const float*)d_in[15], (const float*)d_in[20], (const float*)d_in[25] };
    const float* Wd      = (const float*)d_in[26];
    const float* bd      = (const float*)d_in[27];
    const float* g3      = (const float*)d_in[28];
    const float* be3     = (const float*)d_in[29];

    const int Mlev[4] = { in_sizes[0] / 3, in_sizes[1] / 3, in_sizes[2] / 3, in_sizes[3] / 3 };
    const int Elev[3] = { in_sizes[4], in_sizes[6], in_sizes[8] };
    const int Cin[3]  = { 1, 128, 256 };
    const int Cout[3] = { 128, 256, 512 };
    const float inv_r[3] = { 10.0f, 5.0f, 1.0f / (2.549489742783178f) }; // 1/0.1, 1/0.2, 1/(sqrt(6)+0.1)

    // ---- bump allocator over d_ws --------------------------------------
    char* p = (char*)d_ws;
    auto alloc = [&](size_t bytes) -> char* {
        char* r = p;
        p += (bytes + 255) & ~(size_t)255;
        return r;
    };

    float* agg[3];
    float* cnt[3];
    float* xbuf[3];   // post-conv activations per level
    for (int i = 0; i < 3; ++i) {
        agg[i]  = (float*)alloc((size_t)Mlev[i + 1] * 16 * Cin[i] * sizeof(float));
        cnt[i]  = (float*)alloc((size_t)Mlev[i + 1] * sizeof(float));
        xbuf[i] = (float*)alloc((size_t)Mlev[i + 1] * Cout[i] * sizeof(float));
    }
    // packed-A buffer: max over all GEMMs (Kpad elements per row)
    size_t packAElems = 0;
    {
        size_t s0 = (size_t)Mlev[1] * 32;     // level0: K=16 -> Kpad=32
        size_t s1 = (size_t)Mlev[2] * 2048;
        size_t s2 = (size_t)Mlev[3] * 4096;
        size_t s3 = (size_t)Mlev[3] * 512;    // final dense
        packAElems = s0;
        if (s1 > packAElems) packAElems = s1;
        if (s2 > packAElems) packAElems = s2;
        if (s3 > packAElems) packAElems = s3;
    }
    __bf16* packA = (__bf16*)alloc(packAElems * sizeof(__bf16));
    size_t packWElems = (size_t)512 * 4096;   // largest of {128*32, 256*2048, 512*4096, 128*512}
    __bf16* packW = (__bf16*)alloc(packWElems * sizeof(__bf16));
    float* csum   = (float*)alloc(512 * sizeof(float));
    float* csumsq = (float*)alloc(512 * sizeof(float));

    const int BT = 256;

    // ---- 3 Monte-Carlo conv levels -------------------------------------
    const float* feat = feat0;
    for (int i = 0; i < 3; ++i) {
        const int E = Elev[i], M = Mlev[i + 1];
        const int K = 16 * Cin[i], Kpad = (K + 31) & ~31, N = Cout[i];

        hipMemsetAsync(agg[i], 0, (size_t)M * K * sizeof(float), stream);
        hipMemsetAsync(cnt[i], 0, (size_t)M * sizeof(float), stream);

        if (E > 0) {
            edge_count_kernel<<<cdiv(E, BT), BT, 0, stream>>>(dstm[i], cnt[i], E);
            edge_scatter_kernel<<<cdiv((long long)E * Cin[i], BT), BT, 0, stream>>>(
                feat, pos[i], pos[i + 1], src[i], dstm[i], Wb[i], bb[i],
                inv_r[i], E, Cin[i], agg[i]);
        }

        norm_pack_kernel<<<cdiv((long long)M * Kpad, BT), BT, 0, stream>>>(
            agg[i], cnt[i], packA, M, K, Kpad);
        pack_wt_kernel<<<cdiv((long long)N * Kpad, BT), BT, 0, stream>>>(
            Wc[i], packW, K, N, Kpad);

        int tiles = ((M + 15) / 16) * (N / 16);
        gemm_wmma_kernel<<<cdiv(tiles, 8), BT, 0, stream>>>(
            packA, packW, nullptr, xbuf[i], M, N, Kpad);

        col_stats_kernel<<<N, BT, 0, stream>>>(xbuf[i], M, N, csum, csumsq);
        bn_lrelu_kernel<<<cdiv((long long)M * N, BT), BT, 0, stream>>>(
            xbuf[i], M, N, csum, csumsq, gg[i], be[i]);

        feat = xbuf[i];
    }

    // ---- final dense (512 -> 128) + BN + lrelu, written to d_out -------
    {
        const int M = Mlev[3], K = 512, Kpad = 512, N = 128;
        float* out = (float*)d_out;

        norm_pack_kernel<<<cdiv((long long)M * Kpad, BT), BT, 0, stream>>>(
            xbuf[2], nullptr, packA, M, K, Kpad);
        pack_wt_kernel<<<cdiv((long long)N * Kpad, BT), BT, 0, stream>>>(
            Wd, packW, K, N, Kpad);

        int tiles = ((M + 15) / 16) * (N / 16);
        gemm_wmma_kernel<<<cdiv(tiles, 8), BT, 0, stream>>>(
            packA, packW, bd, out, M, N, Kpad);

        col_stats_kernel<<<N, BT, 0, stream>>>(out, M, N, csum, csumsq);
        bn_lrelu_kernel<<<cdiv((long long)M * N, BT), BT, 0, stream>>>(
            out, M, N, csum, csumsq, g3, be3);
    }
}
